// S2MoELinear_23356032156126
// MI455X (gfx1250) — compile-verified, matrix operand order checked by reading, not run on hardware
//
#include <hip/hip_runtime.h>
#include <stdint.h>

#define DEVFN __device__ __forceinline__

typedef __attribute__((ext_vector_type(16))) __bf16 v16bf;
typedef __attribute__((ext_vector_type(8)))  float  v8f;

union FragU { struct { uint4 lo, hi; } u; v16bf v; };
static_assert(sizeof(FragU) == 32, "frag size");

constexpr int kN    = 8192;   // B*S
constexpr int kDin  = 4096;
constexpr int kDout = 4096;
constexpr int kE    = 8;
constexpr int kR    = 512;

// ---------- bf16 helpers (round-to-nearest-even) ----------
DEVFN unsigned pk_bf16(float a, float b) {
  unsigned ua = __builtin_bit_cast(unsigned, a);
  unsigned ub = __builtin_bit_cast(unsigned, b);
  ua += 0x7FFFu + ((ua >> 16) & 1u);
  ub += 0x7FFFu + ((ub >> 16) & 1u);
  return (ua >> 16) | (ub & 0xFFFF0000u);
}
DEVFN unsigned short bf16_1(float a) {
  unsigned ua = __builtin_bit_cast(unsigned, a);
  ua += 0x7FFFu + ((ua >> 16) & 1u);
  return (unsigned short)(ua >> 16);
}

DEVFN v8f wmma_bf16(v16bf a, v16bf b, v8f c) {
  // v_wmma_f32_16x16x32_bf16 : D = A(16x32) * B(32x16) + C(16x16 f32)
  return __builtin_amdgcn_wmma_f32_16x16x32_bf16(false, a, false, b, (short)0, c, false, false);
}

DEVFN v8f zero_v8f() { v8f z = {0.f,0.f,0.f,0.f,0.f,0.f,0.f,0.f}; return z; }

// 16x32 fragment from row-major bf16 global memory: 16 rows, K contiguous.
// Lane L (L<16): row=L, K={0..7} and {16..23}; lane L+16: row=L, K={8..15},{24..31}.
DEVFN v16bf frag_bf16_g(const unsigned short* __restrict__ base, int row0, int ld,
                        int k0, int lane) {
  int half = lane >> 4, m = lane & 15;
  const unsigned short* p = base + (size_t)(row0 + m) * ld + k0 + half * 8;
  FragU f;
  f.u.lo = *(const uint4*)(p);
  f.u.hi = *(const uint4*)(p + 16);
  return f.v;
}

// 16x32 fragment from padded LDS bf16 tile, row stride 40 elems (80 B => 16B-aligned,
// 20-DWORD bank stride: conflict-free b128 reads across 16 rows).
DEVFN v16bf frag_lds(const unsigned short (*w)[40], int lane) {
  int half = lane >> 4, m = lane & 15;
  const unsigned short* p = &w[m][half * 8];
  FragU f;
  f.u.lo = *(const uint4*)(p);
  f.u.hi = *(const uint4*)(p + 16);
  return f.v;
}

// ---------------- conversion kernels (one-time f32 -> bf16 passes) ----------------
__global__ __launch_bounds__(256) void k_cvt_flat(const float* __restrict__ src,
                                                  unsigned short* __restrict__ dst,
                                                  size_t n) {
  size_t i = ((size_t)blockIdx.x * 256 + threadIdx.x) * 4;
  if (i < n) {
    float4 v = *(const float4*)(src + i);
    uint2 r; r.x = pk_bf16(v.x, v.y); r.y = pk_bf16(v.z, v.w);
    *(uint2*)(dst + i) = r;
  }
}

// Wb[m][o][d]: m=0 -> pretrained_w, m=1..8 -> expert_w[m-1]
__global__ __launch_bounds__(256) void k_cvt_w(const float* __restrict__ pw,
                                               const float* __restrict__ ew,
                                               unsigned short* __restrict__ wb) {
  constexpr size_t PWN = (size_t)kDout * kDin;
  size_t i = ((size_t)blockIdx.x * 256 + threadIdx.x) * 4;
  if (i < 9 * PWN) {
    const float* src = (i < PWN) ? (pw + i) : (ew + (i - PWN));
    float4 v = *(const float4*)src;
    uint2 r; r.x = pk_bf16(v.x, v.y); r.y = pk_bf16(v.z, v.w);
    *(uint2*)(wb + i) = r;
  }
}

// Vt[e][r][d] = gate[e][d][r], bf16.  64x64 tiles through LDS for coalescing.
__global__ __launch_bounds__(256) void k_cvt_gate_t(const float* __restrict__ gate,
                                                    unsigned short* __restrict__ vt) {
  __shared__ __align__(8) unsigned short tl[64][72];
  int tid = threadIdx.x;
  int r0 = blockIdx.x * 64, d0 = blockIdx.y * 64, e = blockIdx.z;
  const float* V = gate + (size_t)e * kDin * kR;
#pragma unroll
  for (int t = 0; t < 4; ++t) {
    int q = tid + t * 256;            // 0..1023 float4-units of a 64x64 f32 tile
    int row = q >> 4;                 // d within tile
    int rq  = q & 15;                 // float4 along r
    float4 v = *(const float4*)(V + (size_t)(d0 + row) * kR + r0 + rq * 4);
    unsigned* p = (unsigned*)&tl[row][rq * 4];
    p[0] = pk_bf16(v.x, v.y);
    p[1] = pk_bf16(v.z, v.w);
  }
  __syncthreads();
#pragma unroll
  for (int t = 0; t < 4; ++t) {
    int q = tid + t * 256;            // 0..1023 uint2-units (4 bf16) of transposed tile
    int row = q >> 4;                 // r within tile
    int dq  = q & 15;                 // 4-elem group along d
    unsigned u0 = tl[dq * 4 + 0][row];
    unsigned u1 = tl[dq * 4 + 1][row];
    unsigned u2 = tl[dq * 4 + 2][row];
    unsigned u3 = tl[dq * 4 + 3][row];
    uint2 r; r.x = (u1 << 16) | u0; r.y = (u3 << 16) | u2;
    *(uint2*)(vt + ((size_t)e * kR + r0 + row) * kDin + d0 + dq * 4) = r;
  }
}

// ---------------- kernel 1: T[e] = Xb @ V_e  (bf16 out) ----------------
// Pure streaming: A from xb, B from pre-transposed Vt — both K-contiguous, no LDS.
// block: 256 thr = 8 waves (2 M x 4 N) -> 32x64 tile of T. grid: (N/32, R/64, E)
__global__ __launch_bounds__(256) void k_gate_proj(const unsigned short* __restrict__ xb,
                                                   const unsigned short* __restrict__ vt,
                                                   unsigned short* __restrict__ T) {
  int tid = threadIdx.x, lane = tid & 31, wave = tid >> 5;
  int wm = (wave >> 2) * 16, wn = (wave & 3) * 16;
  int n0 = blockIdx.x * 32, r0 = blockIdx.y * 64, e = blockIdx.z;
  const unsigned short* Vt = vt + (size_t)e * kR * kDin;   // [R, Din] bf16
  v8f acc = zero_v8f();
  for (int k0 = 0; k0 < kDin; k0 += 32) {
    v16bf a = frag_bf16_g(xb, n0 + wm, kDin, k0, lane);
    v16bf b = frag_bf16_g(Vt, r0 + wn, kDin, k0, lane);   // B[k,r] = V[k,r]
    acc = wmma_bf16(a, b, acc);
  }
  int half = lane >> 4;
  int rg = r0 + wn + (lane & 15);
#pragma unroll
  for (int r = 0; r < 8; ++r) {
    int n = n0 + wm + r + 8 * half;
    T[((size_t)e * kN + n) * kR + rg] = bf16_1(acc[r]);
  }
}

// ---------------- kernel 2: res[n,e] = || x - T_e @ V_e^T || ----------------
// block: 256 thr, 16 tokens; wave w sweeps d-tiles d = dt*128 + w*16. grid: (N/16, E)
__global__ __launch_bounds__(256) void k_residual(const float* __restrict__ x,
                                                  const unsigned short* __restrict__ vb,
                                                  const unsigned short* __restrict__ T,
                                                  float* __restrict__ res) {
  __shared__ float part[8][16];
  int tid = threadIdx.x, lane = tid & 31, wave = tid >> 5;
  int half = lane >> 4;
  int n0 = blockIdx.x * 16, e = blockIdx.y;
  const unsigned short* V = vb + (size_t)e * kDin * kR;        // [Din, R] bf16
  const unsigned short* Tb = T + ((size_t)e * kN + n0) * kR;   // [16, R] bf16
  float rs[8] = {0.f,0.f,0.f,0.f,0.f,0.f,0.f,0.f};
  for (int dt = 0; dt < kDin / 128; ++dt) {
    int d0 = dt * 128 + wave * 16;
    v8f acc = zero_v8f();
    for (int r0 = 0; r0 < kR; r0 += 32) {
      v16bf a = frag_bf16_g(Tb, 0, kR, r0, lane);      // A: T tile, K=r contiguous
      v16bf b = frag_bf16_g(V, d0, kR, r0, lane);      // B[r,d] = V[d,r], contiguous in r
      acc = wmma_bf16(a, b, acc);
    }
    int d = d0 + (lane & 15);
#pragma unroll
    for (int r = 0; r < 8; ++r) {
      int m = r + 8 * half;
      float xv = x[(size_t)(n0 + m) * kDin + d];
      float df = xv - acc[r];
      rs[r] += df * df;
    }
  }
  // deterministic cross-lane reduction over the 16 columns of each half
#pragma unroll
  for (int r = 0; r < 8; ++r) {
    rs[r] += __shfl_xor(rs[r], 1);
    rs[r] += __shfl_xor(rs[r], 2);
    rs[r] += __shfl_xor(rs[r], 4);
    rs[r] += __shfl_xor(rs[r], 8);
  }
  if ((lane & 15) == 0) {
#pragma unroll
    for (int r = 0; r < 8; ++r) part[wave][r + 8 * half] = rs[r];
  }
  __syncthreads();
  if (tid < 16) {
    float s = 0.f;
#pragma unroll
    for (int w = 0; w < 8; ++w) s += part[w][tid];
    res[(size_t)(n0 + tid) * kE + e] = sqrtf(s);
  }
}

// ---------------- kernel 3: softmax(-res), threshold 1/E, top-2, normalize ----------------
__global__ __launch_bounds__(256) void k_route(const float* __restrict__ res,
                                               float* __restrict__ nw) {
  int t = blockIdx.x * 256 + threadIdx.x;
  if (t >= kN) return;
  float rw[kE];
  float mx = -1e30f;
#pragma unroll
  for (int e = 0; e < kE; ++e) { rw[e] = -res[(size_t)t * kE + e]; mx = fmaxf(mx, rw[e]); }
  float s = 0.f;
#pragma unroll
  for (int e = 0; e < kE; ++e) { rw[e] = __expf(rw[e] - mx); s += rw[e]; }
  float inv = 1.f / s;
#pragma unroll
  for (int e = 0; e < kE; ++e) rw[e] *= inv;
  int i1 = 0;
#pragma unroll
  for (int e = 1; e < kE; ++e) if (rw[e] > rw[i1]) i1 = e;
  int i2 = (i1 == 0) ? 1 : 0;
#pragma unroll
  for (int e = 0; e < kE; ++e) if (e != i1 && rw[e] > rw[i2]) i2 = e;
  float fw[kE]; float fs = 0.f;
#pragma unroll
  for (int e = 0; e < kE; ++e) {
    float f = ((e == i1 || e == i2) && rw[e] > (1.0f / kE)) ? rw[e] : 0.f;
    fw[e] = f; fs += f;
  }
  if (fs == 0.f) fs = 1.f;
  float r = 1.f / fs;
#pragma unroll
  for (int e = 0; e < kE; ++e) nw[(size_t)t * kE + e] = fw[e] * r;
}

// ---------------- kernel 4: fused pretrained + 8-expert GEMM + combine ----------------
// block: 256 thr = 8 waves (2 M x 4 N) -> 32x64 output tile, 9 accumulators per wave.
// Double-buffered async pipeline: chunk i+1 streams into LDS buffer B(i+1) via
// GLOBAL_LOAD_ASYNC_TO_LDS_B128 while the 9 WMMAs of chunk i consume buffer B(i).
// ASYNCcnt is a per-wave FIFO (9 ops/chunk), so s_wait_asynccnt 9 after issuing the
// next chunk guarantees the current chunk has landed.
// grid: (N/32 fastest, Dout/64) so the ~4.7MB bf16 weight column-panel stays hot in L2.
__global__ __launch_bounds__(256) void k_moe_main(const unsigned short* __restrict__ xb,
                                                  const unsigned short* __restrict__ wb,
                                                  const float* __restrict__ pb,
                                                  const float* __restrict__ eb,
                                                  const float* __restrict__ nw,
                                                  float* __restrict__ out) {
  __shared__ __align__(16) unsigned short wlds[2][9][64][40];   // 2 x 46 KB ping-pong
  int tid = threadIdx.x, lane = tid & 31, wave = tid >> 5;
  int wm = (wave >> 2) * 16, wn = (wave & 3) * 16;
  int n0 = blockIdx.x * 32, o0 = blockIdx.y * 64;
  int half = lane >> 4;
  unsigned ldsbase = (unsigned)(uintptr_t)&wlds[0][0][0][0];
  constexpr unsigned kBufBytes = 9u * 64u * 40u * 2u;           // 46080

  // stage one 9x[64x32] bf16 weight chunk into LDS buffer `buf` (9 x b128 per thread)
  auto issue_chunk = [&](int K0, int buf) {
#pragma unroll
    for (int t = 0; t < 9; ++t) {
      int id  = tid + t * 256;          // 0..2303
      int m   = id >> 8;                // matrix 0..8
      int rem = id & 255;
      int row = rem >> 2;               // 0..63
      int q   = rem & 3;                // 16B quarter within the 64B row chunk
      const unsigned short* src = wb + ((size_t)m * kDout + o0 + row) * kDin + K0 + q * 8;
      unsigned dst = ldsbase + (unsigned)buf * kBufBytes
                   + (unsigned)((((m * 64 + row) * 40) + q * 8) * 2);
      asm volatile("global_load_async_to_lds_b128 %0, %1, off"
                   :: "v"(dst), "v"(src) : "memory");
    }
  };

  v8f acc[9];
#pragma unroll
  for (int m = 0; m < 9; ++m) acc[m] = zero_v8f();

  issue_chunk(0, 0);                      // prologue: chunk 0 -> buffer 0
  for (int k0 = 0, it = 0; k0 < kDin; k0 += 32, ++it) {
    int cur = it & 1;
    if (k0 + 32 < kDin) {
      issue_chunk(k0 + 32, cur ^ 1);      // overlap next chunk's copy with compute
      asm volatile("s_wait_asynccnt 9" ::: "memory");   // chunk `it` landed
    } else {
      asm volatile("s_wait_asynccnt 0" ::: "memory");
    }
    __syncthreads();                      // cross-wave visibility of buffer `cur`
    v16bf a = frag_bf16_g(xb, n0 + wm, kDin, k0, lane);        // shared A fragment
#pragma unroll
    for (int m = 0; m < 9; ++m) {
      v16bf b = frag_lds(&wlds[cur][m][wn], lane);
      acc[m] = wmma_bf16(a, b, acc[m]);
    }
    __syncthreads();                      // all reads of `cur` done before it is refilled
  }

  // epilogue: out = acc_p + pb + sum_e nw[n,e]*(acc_e + eb[e])
  int o = o0 + wn + (lane & 15);
  float pbv = pb[o];
  float ebv[8];
#pragma unroll
  for (int e2 = 0; e2 < 8; ++e2) ebv[e2] = eb[(size_t)e2 * kDout + o];
#pragma unroll
  for (int r = 0; r < 8; ++r) {
    int n = n0 + wm + r + 8 * half;
    const float* nwr = nw + (size_t)n * kE;
    float v = acc[0][r] + pbv;
#pragma unroll
    for (int e2 = 0; e2 < 8; ++e2) v += nwr[e2] * (acc[e2 + 1][r] + ebv[e2]);
    out[(size_t)n * kDout + o] = v;
  }
}

// ---------------- host launcher ----------------
extern "C" void kernel_launch(void* const* d_in, const int* in_sizes, int n_in,
                              void* d_out, int out_size, void* d_ws, size_t ws_size,
                              hipStream_t stream) {
  (void)in_sizes; (void)n_in; (void)out_size; (void)ws_size;
  const float* x    = (const float*)d_in[0];   // [B,S,Din]
  const float* gate = (const float*)d_in[1];   // [E,Din,R]
  const float* pw   = (const float*)d_in[2];   // [Dout,Din]
  const float* pb   = (const float*)d_in[3];   // [Dout]
  const float* ew   = (const float*)d_in[4];   // [E,Dout,Din]
  const float* eb   = (const float*)d_in[5];   // [E,Dout]
  float* out = (float*)d_out;

  // workspace layout (bytes)
  constexpr size_t XB_B = (size_t)kN * kDin * 2;            //  64 MB bf16 x
  constexpr size_t T_B  = (size_t)kE * kN * kR * 2;         //  64 MB bf16 T
  constexpr size_t WB_B = (size_t)9 * kDout * kDin * 2;     // 288 MB bf16 weights
  constexpr size_t VB_B = (size_t)kE * kDin * kR * 2;       //  32 MB bf16 gate
  constexpr size_t VT_B = VB_B;                             //  32 MB bf16 gate^T
  char* w = (char*)d_ws;
  unsigned short* xb = (unsigned short*)(w);
  unsigned short* T  = (unsigned short*)(w + XB_B);
  unsigned short* wb = (unsigned short*)(w + XB_B + T_B);
  unsigned short* vb = (unsigned short*)(w + XB_B + T_B + WB_B);
  unsigned short* vt = (unsigned short*)(w + XB_B + T_B + WB_B + VB_B);
  float* res = (float*)(w + XB_B + T_B + WB_B + VB_B + VT_B);
  float* nw  = res + (size_t)kN * kE;

  const size_t nX = (size_t)kN * kDin;
  const size_t nV = (size_t)kE * kDin * kR;
  const size_t nW = (size_t)9 * kDout * kDin;

  k_cvt_flat  <<<dim3((unsigned)(nX / 4 / 256)), 256, 0, stream>>>(x, xb, nX);
  k_cvt_w     <<<dim3((unsigned)(nW / 4 / 256)), 256, 0, stream>>>(pw, ew, wb);
  k_cvt_flat  <<<dim3((unsigned)(nV / 4 / 256)), 256, 0, stream>>>(gate, vb, nV);
  k_cvt_gate_t<<<dim3(kR / 64, kDin / 64, kE), 256, 0, stream>>>(gate, vt);
  k_gate_proj <<<dim3(kN / 32, kR / 64, kE), 256, 0, stream>>>(xb, vt, T);
  k_residual  <<<dim3(kN / 16, kE), 256, 0, stream>>>(x, vb, T, res);
  k_route     <<<dim3((kN + 255) / 256), 256, 0, stream>>>(res, nw);
  k_moe_main  <<<dim3(kN / 32, kDout / 64), 256, 0, stream>>>(xb, wb, pb, eb, nw, out);
}
